// Net_22643067584549
// MI455X (gfx1250) — matrix operand order repaired and checked by reading, not compiled, not run on hardware
//
#include <hip/hip_runtime.h>
#include <hip/hip_bf16.h>
#include <math.h>

// ---------------------------------------------------------------------------
// DGCNN / EdgeConv for MI455X (gfx1250, wave32, WMMA).
//
// EdgeConv refactor:  max_j th(x_j - x_i) + phi(x_i)
//                  =  (max_j Yt[j]) - Yt[i] + tb + Yp[i] + pb
// with Yt = X @ th^T, Yp = X @ phi^T  -> dense f16 WMMA GEMMs + gather-max.
//
// GEMM is register-blocked 16x64 per wave: one A fragment feeds 4 WMMAs per
// K-step (2 A loads + 8 B loads -> 4 v_wmma_f32_16x16x32_f16).
// ---------------------------------------------------------------------------

typedef _Float16 v16h __attribute__((ext_vector_type(16)));
typedef _Float16 v8h  __attribute__((ext_vector_type(8)));
typedef float    v8f  __attribute__((ext_vector_type(8)));

#define NNODES 38400   // 64 graphs * 600 nodes
#define KNBR   8

// ---------------------------------------------------------------------------
// f32 [rows, C] -> f16 [rowsPad, Cpad], zero padded in both dims.
// ---------------------------------------------------------------------------
__global__ void k_cvt_pad2_f16(const float* __restrict__ src,
                               _Float16* __restrict__ dst,
                               int rows, int C, int rowsPad, int Cpad) {
    int idx = blockIdx.x * blockDim.x + threadIdx.x;
    int total = rowsPad * Cpad;
    if (idx >= total) return;
    int c = idx % Cpad;
    int r = idx / Cpad;
    dst[idx] = (r < rows && c < C) ? (_Float16)src[r * C + c] : (_Float16)0.0f;
}

// ---------------------------------------------------------------------------
// C[M, Npad] = A[M, Kpad](f16) * W[Npad, Kpad](f16)^T, f32 accumulate.
// 256 threads = 8 waves; wave w -> M-tile (blockIdx.x*8 + w),
// N-block = blockIdx.y * 64 (4 register-blocked 16x16 tiles per wave).
// WMMA register layouts used here (A/B in contiguous k order, same for both):
//   A 16x32 : lane l row=l&15 ; K-group (l>>4)*16, 16 contiguous halfs
//   B 32x16 : lane l col=l&15 ; K-group (l>>4)*16, 16 contiguous halfs
//   D 16x16 : vgpr r, lane l -> row r + 8*(l>=16), col l&15
// Kpad % 32 == 0, Npad % 64 == 0, M % 128 == 0 (M = 38400).
// ---------------------------------------------------------------------------
__global__ void __launch_bounds__(256)
k_gemm_f16_wmma(const _Float16* __restrict__ A,
                const _Float16* __restrict__ W,
                float* __restrict__ C,
                int Kpad, int Npad) {
    const int lane  = threadIdx.x & 31;
    const int wave  = threadIdx.x >> 5;
    const int tileM = blockIdx.x * 8 + wave;
    const int nBase = blockIdx.y * 64;
    const int hi    = lane >> 4;       // 0 or 1 (lane group)
    const int l15   = lane & 15;

    const _Float16* aBase = A + (tileM * 16 + l15) * Kpad + hi * 16;  // 32B aligned
    const _Float16* bBase = W + (nBase + l15) * Kpad + hi * 16;       // 32B aligned
    const int bStride = 16 * Kpad;     // halfs between consecutive N-tiles

    v8f acc0 = {}, acc1 = {}, acc2 = {}, acc3 = {};
    for (int k0 = 0; k0 < Kpad; k0 += 32) {
        v8h a0 = *(const v8h*)(aBase + k0);        // K = kb + 0..7
        v8h a1 = *(const v8h*)(aBase + k0 + 8);    // K = kb + 8..15
        v16h a;
#pragma unroll
        for (int t = 0; t < 8; ++t) { a[t] = a0[t]; a[t + 8] = a1[t]; }

        v16h b0 = *(const v16h*)(bBase + k0);
        v16h b1 = *(const v16h*)(bBase + bStride     + k0);
        v16h b2 = *(const v16h*)(bBase + bStride * 2 + k0);
        v16h b3 = *(const v16h*)(bBase + bStride * 3 + k0);

        acc0 = __builtin_amdgcn_wmma_f32_16x16x32_f16(false, a, false, b0,
                                                      (short)0, acc0, false, false);
        acc1 = __builtin_amdgcn_wmma_f32_16x16x32_f16(false, a, false, b1,
                                                      (short)0, acc1, false, false);
        acc2 = __builtin_amdgcn_wmma_f32_16x16x32_f16(false, a, false, b2,
                                                      (short)0, acc2, false, false);
        acc3 = __builtin_amdgcn_wmma_f32_16x16x32_f16(false, a, false, b3,
                                                      (short)0, acc3, false, false);
    }

    float* cBase = C + (tileM * 16 + hi * 8) * Npad + nBase + l15;
#pragma unroll
    for (int r = 0; r < 8; ++r) {
        cBase[r * Npad]      = acc0[r];
        cBase[r * Npad + 16] = acc1[r];
        cBase[r * Npad + 32] = acc2[r];
        cBase[r * Npad + 48] = acc3[r];
    }
}

// ---------------------------------------------------------------------------
// h[i,c] = tanh( max_k Yt[nbr[i,k],c] - Yt[i,c] + tb[c] + Yp[i,c] + pb[c] )
// Output as zero-padded f16 [N, CnextPad] feeding the next GEMM.
// ---------------------------------------------------------------------------
__global__ void k_combine_tanh(const float* __restrict__ Yt,
                               const float* __restrict__ Yp,
                               const int*   __restrict__ nbr,
                               const float* __restrict__ tb,
                               const float* __restrict__ pb,
                               _Float16* __restrict__ H,
                               int Cout, int CoutPad, int CnextPad) {
    int idx = blockIdx.x * blockDim.x + threadIdx.x;
    int total = NNODES * CnextPad;
    if (idx >= total) return;
    int c = idx % CnextPad;
    int i = idx / CnextPad;
    _Float16 outv = (_Float16)0.0f;
    if (c < Cout) {
        const int* nb = nbr + i * KNBR;
        float m = -3.402823e38f;
#pragma unroll
        for (int k = 0; k < KNBR; ++k)
            m = fmaxf(m, Yt[nb[k] * CoutPad + c]);
        float val = m - Yt[i * CoutPad + c] + tb[c] + Yp[i * CoutPad + c] + pb[c];
        outv = (_Float16)tanhf(val);
    }
    H[idx] = outv;
}

// ---------------------------------------------------------------------------
// Layer 3 combine fused with per-node channel max (h3 never materialized):
// g[i] = max_c ( max_k Yt[nbr[i,k],c] - Yt[i,c] + tb[c] + Yp[i,c] + pb[c] )
// One 256-thread block per node; LDS tree reduce.
// ---------------------------------------------------------------------------
__global__ void __launch_bounds__(256)
k_combine3_gmax(const float* __restrict__ Yt,
                const float* __restrict__ Yp,
                const int*   __restrict__ nbr,
                const float* __restrict__ tb,
                const float* __restrict__ pb,
                float* __restrict__ g,
                int Cout, int CoutPad) {
    const int i = blockIdx.x;
    const int* nb = nbr + i * KNBR;
    int j[KNBR];
#pragma unroll
    for (int k = 0; k < KNBR; ++k) j[k] = nb[k] * CoutPad;

    float best = -3.402823e38f;
    for (int c = threadIdx.x; c < Cout; c += blockDim.x) {
        float m = -3.402823e38f;
#pragma unroll
        for (int k = 0; k < KNBR; ++k)
            m = fmaxf(m, Yt[j[k] + c]);
        float val = m - Yt[i * CoutPad + c] + tb[c] + Yp[i * CoutPad + c] + pb[c];
        best = fmaxf(best, val);
    }
    __shared__ float red[256];
    red[threadIdx.x] = best;
    __syncthreads();
#pragma unroll
    for (int s = 128; s > 0; s >>= 1) {
        if (threadIdx.x < s)
            red[threadIdx.x] = fmaxf(red[threadIdx.x], red[threadIdx.x + s]);
        __syncthreads();
    }
    if (threadIdx.x == 0) g[i] = red[0];
}

// ---------------------------------------------------------------------------
// Tiny FC head (64 rows): out[r,c] = act( X[r,:] . W[c,:] + b[c] )
// mode 0 -> tanh, mode 1 -> clip(-2, 2).
// ---------------------------------------------------------------------------
__global__ void k_fc(const float* __restrict__ X,
                     const float* __restrict__ W,
                     const float* __restrict__ b,
                     float* __restrict__ Y,
                     int rows, int Cin, int Cout, int mode) {
    int idx = blockIdx.x * blockDim.x + threadIdx.x;
    if (idx >= rows * Cout) return;
    int r = idx / Cout, c = idx % Cout;
    const float* x = X + r * Cin;
    const float* w = W + c * Cin;
    float s = b[c];
    for (int k = 0; k < Cin; ++k) s += x[k] * w[k];
    Y[idx] = (mode == 0) ? tanhf(s) : fminf(2.0f, fmaxf(-2.0f, s));
}

// ---------------------------------------------------------------------------
// Host orchestration
// ---------------------------------------------------------------------------
static inline int ceil_div(int a, int b) { return (a + b - 1) / b; }

extern "C" void kernel_launch(void* const* d_in, const int* in_sizes, int n_in,
                              void* d_out, int out_size, void* d_ws, size_t ws_size,
                              hipStream_t stream) {
    (void)in_sizes; (void)n_in; (void)out_size; (void)ws_size;

    const float* x    = (const float*)d_in[0];   // [64,600,50]
    const int*   nbr  = (const int*)  d_in[1];   // [N,8]
    const float* t1w  = (const float*)d_in[2];  const float* t1b = (const float*)d_in[3];
    const float* p1w  = (const float*)d_in[4];  const float* p1b = (const float*)d_in[5];
    const float* t2w  = (const float*)d_in[6];  const float* t2b = (const float*)d_in[7];
    const float* p2w  = (const float*)d_in[8];  const float* p2b = (const float*)d_in[9];
    const float* t3w  = (const float*)d_in[10]; const float* t3b = (const float*)d_in[11];
    const float* p3w  = (const float*)d_in[12]; const float* p3b = (const float*)d_in[13];
    const float* fc1w = (const float*)d_in[14]; const float* fc1b = (const float*)d_in[15];
    const float* fc2w = (const float*)d_in[16]; const float* fc2b = (const float*)d_in[17];
    const float* fc3w = (const float*)d_in[18]; const float* fc3b = (const float*)d_in[19];
    const float* fc4w = (const float*)d_in[20]; const float* fc4b = (const float*)d_in[21];
    const float* fow  = (const float*)d_in[22]; const float* fob  = (const float*)d_in[23];
    float* out = (float*)d_out;                  // [64,9]

    // Padded dims: K -> mult of 32, Cout -> mult of 64 (for 16x64 wave tiles).
    // L1: 50->64 in, 100->128 out ; L2: 100->128 in, 200->256 out
    // L3: 200->224 in, 600->640 out
    const int N = NNODES;

    // --- workspace arena ---
    char* ws = (char*)d_ws;
    size_t off = 0;
    auto alloc = [&](size_t bytes) -> void* {
        off = (off + 255) & ~(size_t)255;
        void* p = (void*)(ws + off);
        off += bytes;
        return p;
    };
    _Float16* xpad = (_Float16*)alloc((size_t)N * 64  * 2);
    _Float16* h1   = (_Float16*)alloc((size_t)N * 128 * 2);
    _Float16* h2   = (_Float16*)alloc((size_t)N * 224 * 2);
    float*    Yt   = (float*)   alloc((size_t)N * 640 * 4);
    float*    Yp   = (float*)   alloc((size_t)N * 640 * 4);
    _Float16* wt1  = (_Float16*)alloc((size_t)128 * 64  * 2);
    _Float16* wp1  = (_Float16*)alloc((size_t)128 * 64  * 2);
    _Float16* wt2  = (_Float16*)alloc((size_t)256 * 128 * 2);
    _Float16* wp2  = (_Float16*)alloc((size_t)256 * 128 * 2);
    _Float16* wt3  = (_Float16*)alloc((size_t)640 * 224 * 2);
    _Float16* wp3  = (_Float16*)alloc((size_t)640 * 224 * 2);
    float*    g    = (float*)   alloc((size_t)64 * 600 * 4);
    float*    a1   = (float*)   alloc((size_t)64 * 300 * 4);
    float*    a2   = (float*)   alloc((size_t)64 * 300 * 4);
    float*    a3   = (float*)   alloc((size_t)64 * 100 * 4);
    float*    a4   = (float*)   alloc((size_t)64 * 50  * 4);

    const int T = 256;
    auto cvt = [&](const float* s, _Float16* d, int rows, int C, int rowsPad, int Cpad) {
        int total = rowsPad * Cpad;
        k_cvt_pad2_f16<<<ceil_div(total, T), T, 0, stream>>>(s, d, rows, C, rowsPad, Cpad);
    };
    // GEMM grid: 2400 M-tiles, 8 waves/block -> 300 blocks in x; y = Npad/64.
    auto gemm = [&](const _Float16* A, const _Float16* W, float* C,
                    int Kpad, int Npad) {
        dim3 grid(N / 128, Npad / 64);
        k_gemm_f16_wmma<<<grid, 256, 0, stream>>>(A, W, C, Kpad, Npad);
    };

    // ---- EdgeConv 1: 50 -> 100 (tanh) ----
    cvt(x,   xpad, N,   50, N,   64);
    cvt(t1w, wt1,  100, 50, 128, 64);
    cvt(p1w, wp1,  100, 50, 128, 64);
    gemm(xpad, wt1, Yt, 64, 128);
    gemm(xpad, wp1, Yp, 64, 128);
    k_combine_tanh<<<ceil_div(N * 128, T), T, 0, stream>>>(
        Yt, Yp, nbr, t1b, p1b, h1, 100, 128, 128);

    // ---- EdgeConv 2: 100 -> 200 (tanh) ----
    cvt(t2w, wt2, 200, 100, 256, 128);
    cvt(p2w, wp2, 200, 100, 256, 128);
    gemm(h1, wt2, Yt, 128, 256);
    gemm(h1, wp2, Yp, 128, 256);
    k_combine_tanh<<<ceil_div(N * 224, T), T, 0, stream>>>(
        Yt, Yp, nbr, t2b, p2b, h2, 200, 256, 224);

    // ---- EdgeConv 3: 200 -> 600 (linear) + fused per-node channel max ----
    cvt(t3w, wt3, 600, 200, 640, 224);
    cvt(p3w, wp3, 600, 200, 640, 224);
    gemm(h2, wt3, Yt, 224, 640);
    gemm(h2, wp3, Yp, 224, 640);
    k_combine3_gmax<<<N, 256, 0, stream>>>(Yt, Yp, nbr, t3b, p3b, g, 600, 640);

    // ---- FC head on g[64, 600] ----
    k_fc<<<ceil_div(64 * 300, T), T, 0, stream>>>(g,  fc1w, fc1b, a1, 64, 600, 300, 0);
    k_fc<<<ceil_div(64 * 300, T), T, 0, stream>>>(a1, fc2w, fc2b, a2, 64, 300, 300, 0);
    k_fc<<<ceil_div(64 * 100, T), T, 0, stream>>>(a2, fc3w, fc3b, a3, 64, 300, 100, 0);
    k_fc<<<ceil_div(64 * 50,  T), T, 0, stream>>>(a3, fc4w, fc4b, a4, 64, 100, 50,  0);
    k_fc<<<ceil_div(64 * 9,   T), T, 0, stream>>>(a4, fow,  fob,  out, 64, 50,  9,  1);
}
